// MemoryBank_74096775791029
// MI455X (gfx1250) — matrix operand order repaired and checked by reading, not compiled, not run on hardware
//
#include <hip/hip_runtime.h>
#include <cstdint>

// ---------------------------------------------------------------------------
// out[q] = min_n || emb[q] - bank[n] || ,  d2 = q_sq + (m_sq - 2 q.m)
// q.m via V_WMMA_F32_16X16X32_BF16 with bf16 hi/lo split (ah*bh+ah*bl+al*bh).
// Fast path: pre-convert fp32 -> packed bf16 hi/lo planes in WMMA fragment
// order in d_ws (one pass); hot loop = double-buffered async-to-LDS copy
// overlapped with WMMA. Fallback (small ws): fused conversion while staging.
// ---------------------------------------------------------------------------

typedef __attribute__((ext_vector_type(16))) __bf16 v16bf;
typedef __attribute__((ext_vector_type(8)))  float  v8f;
typedef __attribute__((ext_vector_type(4)))  int    v4i;

#define DDIM   768
#define KPAN   24        // DDIM / 32 k-panels
#define MBLK   64
#define NBLK   128
#define KSTEP  32
#define NSPLIT 16
#define LANEU  20        // LDS lane stride in uints (80B: 64B data + 16B pad)
#define FRAGU  (32*LANEU) // LDS fragment stride in uints (2560B)

union BOp { v16bf v; unsigned u[8]; };

#if defined(__AMDGCN__) && __has_builtin(__builtin_amdgcn_global_load_async_to_lds_b128)
#define HAVE_ASYNC 1
#else
#define HAVE_ASYNC 0
#endif

// round-to-nearest-even fp32 -> bf16 bits in [31:16]
__device__ __forceinline__ unsigned bfrnd(float x) {
    unsigned u = __float_as_uint(x);
    return (u + 0x7FFFu + ((u >> 16) & 1u)) & 0xFFFF0000u;
}
// two floats -> packed bf16 hi-pair + residual lo-pair
__device__ __forceinline__ void cvt2(float x, float y, unsigned &hi, unsigned &lo) {
    unsigned rx = bfrnd(x), ry = bfrnd(y);
    hi = (rx >> 16) | ry;
    float xr = x - __uint_as_float(rx);
    float yr = y - __uint_as_float(ry);
    unsigned sx = bfrnd(xr), sy = bfrnd(yr);
    lo = (sx >> 16) | sy;
}
// order-preserving float -> uint key
__device__ __forceinline__ unsigned fkey(float f) {
    unsigned u = __float_as_uint(f);
    return (u & 0x80000000u) ? ~u : (u | 0x80000000u);
}

// 16B global -> LDS copy: async-to-LDS when available
__device__ __forceinline__ void copy16(const unsigned* __restrict__ src, unsigned* dst) {
#if HAVE_ASYNC
    __builtin_amdgcn_global_load_async_to_lds_b128(
        (__attribute__((address_space(1))) v4i*)(uintptr_t)src,
        (__attribute__((address_space(3))) v4i*)(unsigned)(uintptr_t)dst,
        0, 0);
#else
    uint4 v = *(const uint4*)src;
    *(uint4*)dst = v;
#endif
}
__device__ __forceinline__ void copyWait() {
#if HAVE_ASYNC
#if __has_builtin(__builtin_amdgcn_s_wait_asynccnt)
    __builtin_amdgcn_s_wait_asynccnt(0);
#else
    asm volatile("s_wait_asynccnt 0x0" ::: "memory");
#endif
#endif
}

// load one 16x32 fragment's operand regs for this lane (hi then lo planes)
__device__ __forceinline__ void loadFrag(const unsigned* p, BOp &hi, BOp &lo) {
    *(uint4*)&hi.u[0] = *(const uint4*)(p);
    *(uint4*)&hi.u[4] = *(const uint4*)(p + 4);
    *(uint4*)&lo.u[0] = *(const uint4*)(p + 8);
    *(uint4*)&lo.u[4] = *(const uint4*)(p + 12);
}

__global__ void initWs(unsigned* __restrict__ ws, int nkeys, int nmsq) {
    int i = blockIdx.x * 256 + threadIdx.x;
    if (i < nkeys) ws[i] = 0xFFFFFFFFu;
    else if (i < nkeys + nmsq) ws[i] = 0u;   // fp32 0.0 for msq accumulators
}

// ---- one-time fp32 -> bf16 hi/lo fragment-order conversion (A matrix) ----
// A frag (16 rows x 32 k): lane = r + 16*(k>>4); v = (k&15)>>1
__global__ __launch_bounds__(256)
void convertA(const float* __restrict__ src, unsigned* __restrict__ dst, int nfrag) {
    int wv = blockIdx.x * 8 + (threadIdx.x >> 5);
    if (wv >= nfrag) return;
    int lane = threadIdx.x & 31;
    int fr = wv / KPAN, kbi = wv % KPAN;
    int r = fr * 16 + (lane & 15);
    int o = (lane & 16) ? 16 : 0;
    const float* p = src + (long)r * DDIM + kbi * 32 + o;
    unsigned out[16];
#pragma unroll
    for (int run = 0; run < 2; ++run)
#pragma unroll
        for (int q = 0; q < 4; ++q) {
            unsigned hi, lo;
            cvt2(p[run * 8 + 2 * q], p[run * 8 + 2 * q + 1], hi, lo);
            out[run * 4 + q] = hi;
            out[8 + run * 4 + q] = lo;
        }
    unsigned* d = dst + (long)wv * 512 + lane * 16;
#pragma unroll
    for (int q = 0; q < 4; ++q)
        *(uint4*)(d + q * 4) = make_uint4(out[q*4], out[q*4+1], out[q*4+2], out[q*4+3]);
}

// ---- B matrix conversion + exact fp32 row sum-of-squares ----
// B frag (16 cols x 32 k): lane = n + 16*(k>>4); v = (k&15)>>1
__global__ __launch_bounds__(256)
void convertB(const float* __restrict__ src, unsigned* __restrict__ dst,
              float* __restrict__ gMsq, int nfrag) {
    int wv = blockIdx.x * 8 + (threadIdx.x >> 5);
    if (wv >= nfrag) return;
    int lane = threadIdx.x & 31;
    int fr = wv / KPAN, kbi = wv % KPAN;
    int n = fr * 16 + (lane & 15);
    int o = (lane & 16) ? 16 : 0;
    const float* p = src + (long)n * DDIM + kbi * 32 + o;
    unsigned out[16];
    float ss = 0.f;
#pragma unroll
    for (int q = 0; q < 8; ++q) {
        float x = p[2 * q], y = p[2 * q + 1];
        ss = fmaf(x, x, fmaf(y, y, ss));
        unsigned hi, lo;
        cvt2(x, y, hi, lo);
        out[q] = hi; out[8 + q] = lo;
    }
    ss += __shfl_xor(ss, 16, 32);
    if (lane < 16) atomicAdd(&gMsq[n], ss);
    unsigned* d = dst + (long)wv * 512 + lane * 16;
#pragma unroll
    for (int q = 0; q < 4; ++q)
        *(uint4*)(d + q * 4) = make_uint4(out[q*4], out[q*4+1], out[q*4+2], out[q*4+3]);
}

// shared WMMA core: 4 N-tiles x 3 split terms
__device__ __forceinline__ void wmmaStep(const unsigned* sA, const unsigned* sB,
                                         int mW, int nW, int lane, v8f acc[4]) {
    BOp aHi, aLo;
    loadFrag(sA + mW * FRAGU + lane * LANEU, aHi, aLo);
#pragma unroll
    for (int tl = 0; tl < 4; ++tl) {
        BOp bHi, bLo;
        loadFrag(sB + (nW * 4 + tl) * FRAGU + lane * LANEU, bHi, bLo);
        acc[tl] = __builtin_amdgcn_wmma_f32_16x16x32_bf16(
            false, aHi.v, false, bHi.v, (short)0, acc[tl], false, false);
        acc[tl] = __builtin_amdgcn_wmma_f32_16x16x32_bf16(
            false, aHi.v, false, bLo.v, (short)0, acc[tl], false, false);
        acc[tl] = __builtin_amdgcn_wmma_f32_16x16x32_bf16(
            false, aLo.v, false, bHi.v, (short)0, acc[tl], false, false);
    }
}

__device__ __forceinline__ void minFinish(const float* sMsq, v8f acc[4],
                                          int nW, int l16, float vmin[8]) {
#pragma unroll
    for (int tl = 0; tl < 4; ++tl) {
        float msq = sMsq[nW * 64 + tl * 16 + l16];
#pragma unroll
        for (int r = 0; r < 8; ++r)
            vmin[r] = fminf(vmin[r], msq - 2.0f * acc[tl][r]);
    }
}

__device__ __forceinline__ void reduceEmit(float vmin[8], unsigned* wsMin,
                                           int mBase, int mW, int l16, bool hiH) {
#pragma unroll
    for (int r = 0; r < 8; ++r) {
        float v = vmin[r];
        v = fminf(v, __shfl_xor(v, 1, 16));
        v = fminf(v, __shfl_xor(v, 2, 16));
        v = fminf(v, __shfl_xor(v, 4, 16));
        v = fminf(v, __shfl_xor(v, 8, 16));
        if (l16 == 0)
            atomicMin(&wsMin[mBase + mW * 16 + r + (hiH ? 8 : 0)], fkey(v));
    }
}

// ======================= fast path: preconverted data =======================
// Double-buffered pipeline: issue async panel k+1, compute panel k (WMMA
// hides the copy latency), then s_wait_asynccnt + barrier.
__global__ __launch_bounds__(256)
void minDistPre(const unsigned* __restrict__ aconv, const unsigned* __restrict__ bconv,
                const float* __restrict__ gMsq, unsigned* __restrict__ wsMin, int Nbank)
{
    __shared__ __align__(16) unsigned sA0[4 * FRAGU], sA1[4 * FRAGU];
    __shared__ __align__(16) unsigned sB0[8 * FRAGU], sB1[8 * FRAGU];
    __shared__ float sMsq[NBLK];

    const int t = threadIdx.x, lane = t & 31, wave = t >> 5;
    const int mW = wave & 3, nW = wave >> 2, l16 = lane & 15;
    const bool hiH = lane >= 16;
    const int  mBase = blockIdx.x * MBLK;
    const long splitBase = (long)blockIdx.y * (Nbank / NSPLIT);
    const int  nIters = (Nbank / NSPLIT) / NBLK;
    const long aFrag0 = (long)(mBase / 16) * KPAN;

    float vmin[8];
#pragma unroll
    for (int r = 0; r < 8; ++r) vmin[r] = 3.4e38f;

    for (int nb = 0; nb < nIters; ++nb) {
        const long nBase = splitBase + (long)nb * NBLK;
        const long bFrag0 = (nBase / 16) * KPAN;
        __syncthreads();
        if (t < NBLK) sMsq[t] = gMsq[nBase + t];

        // per-thread copy slots: 4 A + 8 B fragments, 6 x 16B per thread
        const unsigned* src[6];
        unsigned *dst0[6], *dst1[6];
#pragma unroll
        for (int c = 0; c < 6; ++c) {
            int cid = t + c * 256, fg = cid >> 7, j = cid & 127;
            unsigned lofs = (unsigned)((j >> 2) * LANEU + (j & 3) * 4);
            if (fg < 4) {
                src[c]  = aconv + (aFrag0 + (long)fg * KPAN) * 512 + j * 4;
                dst0[c] = sA0 + fg * FRAGU + lofs;
                dst1[c] = sA1 + fg * FRAGU + lofs;
            } else {
                src[c]  = bconv + (bFrag0 + (long)(fg - 4) * KPAN) * 512 + j * 4;
                dst0[c] = sB0 + (fg - 4) * FRAGU + lofs;
                dst1[c] = sB1 + (fg - 4) * FRAGU + lofs;
            }
        }
        v8f acc[4] = {v8f{}, v8f{}, v8f{}, v8f{}};

        // prologue: stage panel 0 into buffer 0
#pragma unroll
        for (int c = 0; c < 6; ++c) { copy16(src[c], dst0[c]); src[c] += 512; }
        copyWait();
        __syncthreads();

        for (int kb2 = 0; kb2 < KPAN; kb2 += 2) {
            // stage panel kb2+1 -> buf1, compute panel kb2 from buf0
#pragma unroll
            for (int c = 0; c < 6; ++c) { copy16(src[c], dst1[c]); src[c] += 512; }
            wmmaStep(sA0, sB0, mW, nW, lane, acc);
            copyWait();
            __syncthreads();
            // stage panel kb2+2 -> buf0 (if any), compute panel kb2+1 from buf1
            if (kb2 + 2 < KPAN) {
#pragma unroll
                for (int c = 0; c < 6; ++c) { copy16(src[c], dst0[c]); src[c] += 512; }
            }
            wmmaStep(sA1, sB1, mW, nW, lane, acc);
            copyWait();
            __syncthreads();
        }
        minFinish(sMsq, acc, nW, l16, vmin);
    }
    reduceEmit(vmin, wsMin, mBase, mW, l16, hiH);
}

// ============== fallback path: fused conversion during staging ==============
__global__ __launch_bounds__(256)
void minDistFused(const float* __restrict__ emb, const float* __restrict__ bank,
                  unsigned* __restrict__ wsMin, int Nbank)
{
    __shared__ __align__(16) unsigned sA[4 * FRAGU];
    __shared__ __align__(16) unsigned sB[8 * FRAGU];
    __shared__ float sMsq[NBLK];

    const int t = threadIdx.x, lane = t & 31, wave = t >> 5;
    const int mW = wave & 3, nW = wave >> 2, l16 = lane & 15;
    const bool hiH = lane >= 16;
    const int  mBase = blockIdx.x * MBLK;
    const long splitBase = (long)blockIdx.y * (Nbank / NSPLIT);
    const int  nIters = (Nbank / NSPLIT) / NBLK;

    float vmin[8];
#pragma unroll
    for (int r = 0; r < 8; ++r) vmin[r] = 3.4e38f;

    for (int nb = 0; nb < nIters; ++nb) {
        const long nBase = splitBase + (long)nb * NBLK;
        __syncthreads();
        if (t < NBLK) sMsq[t] = 0.f;
        v8f acc[4] = {v8f{}, v8f{}, v8f{}, v8f{}};

        for (int kb = 0; kb < DDIM; kb += KSTEP) {
            __syncthreads();
            // A: 64x32 fp32 -> fragment-layout bf16 hi/lo
#pragma unroll
            for (int i = 0; i < 2; ++i) {
                int idx = t + i * 256, r = idx >> 3, p4 = idx & 7;
                const float* gp = emb + (long)(mBase + r) * DDIM + kb + p4 * 4;
                float4 g = *(const float4*)gp;
                if (kb + KSTEP < DDIM) __builtin_prefetch(gp + KSTEP, 0, 1);
                unsigned h01, l01, h23, l23;
                cvt2(g.x, g.y, h01, l01);
                cvt2(g.z, g.w, h23, l23);
                int grp = p4 >> 1;
                unsigned* base = sA + (r >> 4) * FRAGU
                               + ((r & 15) + ((p4 >> 2) << 4)) * LANEU;
                int vb = (p4 & 3) << 1;
                *(uint2*)(base + vb)     = make_uint2(h01, h23);
                *(uint2*)(base + 8 + vb) = make_uint2(l01, l23);
            }
            // B: 128x32 fp32 -> fragment layout + msq atomics
#pragma unroll
            for (int i = 0; i < 4; ++i) {
                int idx = t + i * 256, n = idx >> 3, p4 = idx & 7;
                const float* gp = bank + (nBase + n) * DDIM + kb + p4 * 4;
                float4 g = *(const float4*)gp;
                if (kb + KSTEP < DDIM) __builtin_prefetch(gp + KSTEP, 0, 1);
                unsigned h01, l01, h23, l23;
                cvt2(g.x, g.y, h01, l01);
                cvt2(g.z, g.w, h23, l23);
                unsigned* base = sB + (n >> 4) * FRAGU
                               + ((n & 15) + ((p4 >> 2) << 4)) * LANEU;
                int vb = (p4 & 3) << 1;
                *(uint2*)(base + vb)     = make_uint2(h01, h23);
                *(uint2*)(base + 8 + vb) = make_uint2(l01, l23);
                atomicAdd(&sMsq[n], g.x*g.x + g.y*g.y + g.z*g.z + g.w*g.w);
            }
            __syncthreads();
            wmmaStep(sA, sB, mW, nW, lane, acc);
        }
        minFinish(sMsq, acc, nW, l16, vmin);
    }
    reduceEmit(vmin, wsMin, mBase, mW, l16, hiH);
}

__global__ __launch_bounds__(256)
void finalizeKernel(const float* __restrict__ emb, const unsigned* __restrict__ ws,
                    float* __restrict__ out, int Q)
{
    int wave = threadIdx.x >> 5, lane = threadIdx.x & 31;
    int q = blockIdx.x * 8 + wave;
    if (q >= Q) return;
    float s = 0.f;
    for (int k = lane; k < DDIM; k += 32) {
        float x = emb[(long)q * DDIM + k];
        s = fmaf(x, x, s);
    }
#pragma unroll
    for (int m = 16; m; m >>= 1) s += __shfl_xor(s, m, 32);
    if (lane == 0) {
        unsigned k = ws[q];
        unsigned u = (k & 0x80000000u) ? (k & 0x7FFFFFFFu) : ~k;
        out[q] = sqrtf(fmaxf(s + __uint_as_float(u), 0.f));
    }
}

extern "C" void kernel_launch(void* const* d_in, const int* in_sizes, int n_in,
                              void* d_out, int out_size, void* d_ws, size_t ws_size,
                              hipStream_t stream) {
    const float* emb  = (const float*)d_in[0];
    const float* bank = (const float*)d_in[1];
    float* out = (float*)d_out;
    unsigned* ws = (unsigned*)d_ws;
    const int Q = in_sizes[0] / DDIM;   // 8192
    const int N = in_sizes[1] / DDIM;   // 65536
    dim3 grid(Q / MBLK, NSPLIT);

    // ws layout (fast path): [keys Q][msq N][aconv Q*768][bconv N*768] uints
    size_t need = ((size_t)Q + N + (size_t)Q * DDIM + (size_t)N * DDIM) * 4;
    if (ws_size >= need) {
        unsigned* keys  = ws;
        float*    gMsq  = (float*)(ws + Q);
        unsigned* aconv = ws + Q + N;
        unsigned* bconv = aconv + (size_t)Q * DDIM;
        initWs<<<(Q + N + 255) / 256, 256, 0, stream>>>(ws, Q, N);
        int nfA = (Q / 16) * KPAN, nfB = (N / 16) * KPAN;
        convertA<<<(nfA + 7) / 8, 256, 0, stream>>>(emb, aconv, nfA);
        convertB<<<(nfB + 7) / 8, 256, 0, stream>>>(bank, bconv, gMsq, nfB);
        minDistPre<<<grid, 256, 0, stream>>>(aconv, bconv, gMsq, keys, N);
    } else {
        initWs<<<(Q + 255) / 256, 256, 0, stream>>>(ws, Q, 0);
        minDistFused<<<grid, 256, 0, stream>>>(emb, bank, ws, N);
    }
    finalizeKernel<<<(Q + 7) / 8, 256, 0, stream>>>(emb, ws, out, Q);
}
